// GraphAttentionLayer_14697378087172
// MI455X (gfx1250) — matrix operand order repaired and checked by reading, not compiled, not run on hardware
//
#include <hip/hip_runtime.h>
#include <hip/hip_bf16.h>

#define B_  2
#define S_  768
#define D_  768
#define H_  8
#define DH_ 96
#define LN_EPS 1e-5f

typedef _Float16 v16h __attribute__((ext_vector_type(16)));
typedef float    v8f  __attribute__((ext_vector_type(8)));
typedef _Float16 h8   __attribute__((ext_vector_type(8)));
typedef _Float16 h4   __attribute__((ext_vector_type(4)));
typedef int      i4v  __attribute__((vector_size(16)));   // matches builtin param type

#define AS_LOCAL __attribute__((address_space(3)))

#if defined(__has_builtin)
#if __has_builtin(__builtin_amdgcn_global_load_async_to_lds_b128)
#define HAVE_ASYNC 1
#endif
#endif
#ifndef HAVE_ASYNC
#define HAVE_ASYNC 0
#endif

#if HAVE_ASYNC
#if __has_builtin(__builtin_amdgcn_s_wait_asynccnt)
#define ASYNC_WAIT() __builtin_amdgcn_s_wait_asynccnt(0)
#else
#define ASYNC_WAIT() asm volatile("s_wait_asynccnt 0x0" ::: "memory")
#endif
#define ASYNC_CP16(gp, lp) \
    __builtin_amdgcn_global_load_async_to_lds_b128((i4v*)(gp), (AS_LOCAL i4v*)(lp), 0, 0)
#else
#define ASYNC_WAIT() ((void)0)
#define ASYNC_CP16(gp, lp) (*(h8*)(lp) = *(const h8*)(gp))
#endif

// ---- WMMA fragment helpers (K-major source rows) -------------------------
// A-matrix 16x32 f16 layout (ISA 7.12.2): lane half h holds K = 8h+{0..7} and 16+8h+{0..7}
__device__ __forceinline__ v16h frag_a(const _Float16* row, int h) {
    h8 lo = *(const h8*)(row + h * 8);
    h8 hi = *(const h8*)(row + h * 8 + 16);
    v16h f;
#pragma unroll
    for (int e = 0; e < 8; ++e) { f[e] = lo[e]; f[e + 8] = hi[e]; }
    return f;
}
// B-matrix 32x16 f16 layout: lane half h holds K = 8h+{0..7} and 16+8h+{0..7} of column n=lane&15
__device__ __forceinline__ v16h frag_b(const _Float16* row, int h) {
    h8 lo = *(const h8*)(row + h * 8);
    h8 hi = *(const h8*)(row + h * 8 + 16);
    v16h f;
#pragma unroll
    for (int e = 0; e < 8; ++e) { f[e] = lo[e]; f[e + 8] = hi[e]; }
    return f;
}
__device__ __forceinline__ v8f wmma_f16(v16h a, v16h b, v8f c) {
    return __builtin_amdgcn_wmma_f32_16x16x32_f16(false, a, false, b, (short)0, c, false, false);
}

// ---- K0: f32 -> f16 conversion -------------------------------------------
__global__ void cvt_f32_f16(const float* __restrict__ src, _Float16* __restrict__ dst, int n) {
    int i = blockIdx.x * 256 + threadIdx.x;
    if (i < n) dst[i] = (_Float16)src[i];
}

// ---- K1: QKV projection GEMM, heads-scatter epilogue ---------------------
// C[m,n] = sum_k Xh[m,k] * W[n,k] + bias[n];  z selects q/k/v
// Tile: M=32 x N=64, K-chunk 64; waves 2(M) x 4(N), 2 WMMA per wave per chunk.
__global__ void __launch_bounds__(256)
gemm_qkv(const _Float16* __restrict__ Xh,
         const _Float16* __restrict__ Wqh, const _Float16* __restrict__ Wkh,
         const _Float16* __restrict__ Wvh,
         const float* __restrict__ bq, const float* __restrict__ bk, const float* __restrict__ bv,
         _Float16* __restrict__ qh, _Float16* __restrict__ kh, _Float16* __restrict__ vTh) {
    __shared__ _Float16 As[32][72];
    __shared__ _Float16 Bs[64][72];
    const int z = blockIdx.z;
    const _Float16* Wh = (z == 0) ? Wqh : (z == 1) ? Wkh : Wvh;
    const float* bias  = (z == 0) ? bq  : (z == 1) ? bk  : bv;
    const int mBase = blockIdx.y * 32;
    const int nBase = blockIdx.x * 64;
    const int tid = threadIdx.x, lane = tid & 31, wave = tid >> 5;
    const int wm = wave >> 2, wn = wave & 3;   // waves: 2(M) x 4(N)
    const int lr = lane & 15, hh = lane >> 4;
    v8f acc = {};
    for (int kk = 0; kk < D_; kk += 64) {
        // Stage A(32x64) + B(64x64): 768 x 16-byte chunks, 3 per thread (async to LDS)
#pragma unroll
        for (int q3 = 0; q3 < 3; ++q3) {
            int c = tid + q3 * 256;
            const _Float16* gp;
            _Float16* lp;
            if (c < 256) {                      // A chunks: 32 rows x 8 chunks
                int r = c >> 3, col = (c & 7) * 8;
                gp = &Xh[(size_t)(mBase + r) * D_ + kk + col];
                lp = &As[r][col];
            } else {                            // B chunks: 64 rows x 8 chunks
                int c2 = c - 256;
                int r = c2 >> 3, col = (c2 & 7) * 8;
                gp = &Wh[(size_t)(nBase + r) * D_ + kk + col];
                lp = &Bs[r][col];
            }
            ASYNC_CP16(gp, lp);
        }
        ASYNC_WAIT();
        __syncthreads();
#pragma unroll
        for (int kz = 0; kz < 64; kz += 32) {
            v16h a = frag_a(&As[wm * 16 + lr][kz], hh);
            v16h b = frag_b(&Bs[wn * 16 + lr][kz], hh);
            acc = wmma_f16(a, b, acc);
        }
        __syncthreads();
    }
    const int n = nBase + wn * 16 + lr;
    const float bn = bias[n];
    const int head = n / DH_, d = n % DH_;
#pragma unroll
    for (int r = 0; r < 8; ++r) {
        int m = mBase + wm * 16 + r + 8 * hh;
        int bb = m / S_, i = m % S_;
        float val = acc[r] + bn;
        if (z == 2) {
            vTh[(size_t)((bb * H_ + head) * DH_ + d) * S_ + i] = (_Float16)val;
        } else {
            _Float16* dst = (z == 0) ? qh : kh;
            dst[(size_t)((bb * H_ + head) * S_ + i) * DH_ + d] = (_Float16)val;
        }
    }
}

// ---- K2: per-(b,h,i) bias coefficients -----------------------------------
__global__ void coeff_kernel(const _Float16* __restrict__ qh,
                             const float* __restrict__ Wd, const float* __restrict__ bd,
                             const float* __restrict__ Wa, const float* __restrict__ ba,
                             float* __restrict__ cd, float* __restrict__ ca,
                             float* __restrict__ cc) {
    int idx = blockIdx.x * 256 + threadIdx.x;
    if (idx >= B_ * H_ * S_) return;
    const _Float16* qr = qh + (size_t)idx * DH_;
    float sd = 0.f, sa = 0.f, sc = 0.f;
    for (int d = 0; d < DH_; ++d) {
        float qv = (float)qr[d];
        sd += qv * Wd[d];
        sa += qv * Wa[d];
        sc += qv * (bd[d] + ba[d]);
    }
    cd[idx] = sd; ca[idx] = sa; cc[idx] = sc;
}

// ---- K3: fused QK^T + distance/angle/mask bias + softmax -----------------
// One workgroup = (b, h, 16-row i-block) over full j=0..767; probs -> d_out
__global__ void __launch_bounds__(256)
attn_scores(const _Float16* __restrict__ qh, const _Float16* __restrict__ kh,
            const float* __restrict__ dist, const float* __restrict__ ang,
            const float* __restrict__ mask,
            const float* __restrict__ cd, const float* __restrict__ ca,
            const float* __restrict__ cc,
            float* __restrict__ probs) {
    __shared__ _Float16 Aq[16][104];
    __shared__ float Ls[16][772];
    __shared__ float red[16][16];
    __shared__ float scd[16], sca[16], scc[16];

    const int b = blockIdx.z, h = blockIdx.y, iB = blockIdx.x * 16;
    const int tid = threadIdx.x, lane = tid & 31, wave = tid >> 5;
    const int lr = lane & 15, hh = lane >> 4;
    const int bh = b * H_ + h;
    const int jBase = wave * 96;           // 8 waves x 96 cols = 768

    // Prefetch the dist/angle row segments this wave will consume (global_prefetch_b8)
#pragma unroll
    for (int r = 0; r < 8; ++r) {
        int gi = iB + r + 8 * hh;
        __builtin_prefetch(&dist[(size_t)(b * S_ + gi) * S_ + jBase + lr * 6], 0, 0);
        __builtin_prefetch(&ang[(size_t)(b * S_ + gi) * S_ + jBase + lr * 6], 0, 0);
    }

    // Stage q tile (16x96 f16 = 192 x 16-byte chunks) asynchronously to LDS
    if (tid < 192) {
        int r = tid / 12, col = (tid % 12) * 8;
        const _Float16* gp = &qh[(size_t)(bh * S_ + iB + r) * DH_ + col];
        ASYNC_CP16(gp, &Aq[r][col]);
    }
    if (tid < 16) {
        int gi = bh * S_ + iB + tid;
        scd[tid] = cd[gi]; sca[tid] = ca[gi]; scc[tid] = cc[gi];
    }
    ASYNC_WAIT();
    __syncthreads();

    v8f acc[6] = {};
    for (int kk = 0; kk < DH_; kk += 32) {
        v16h a = frag_a(&Aq[lr][kk], hh);
#pragma unroll
        for (int t = 0; t < 6; ++t) {
            const _Float16* brow = kh + (size_t)(bh * S_ + jBase + t * 16 + lr) * DH_ + kk;
            v16h bf = frag_b(brow, hh);
            acc[t] = wmma_f16(a, bf, acc[t]);
        }
    }
    const float inv_s = 0.1020620726159658f;   // 1/sqrt(96)
#pragma unroll
    for (int t = 0; t < 6; ++t) {
        int j = jBase + t * 16 + lr;
        float mterm = (1.0f - mask[b * S_ + j]) * -10000.0f;
#pragma unroll
        for (int r = 0; r < 8; ++r) {
            int il = r + 8 * hh;
            int gi = iB + il;
            float dv = dist[(size_t)(b * S_ + gi) * S_ + j];
            float av = ang[(size_t)(b * S_ + gi) * S_ + j];
            Ls[il][j] = acc[t][r] * inv_s + dv * scd[il] + av * sca[il] + scc[il] + mterm;
        }
    }
    __syncthreads();

    // softmax over j: 16 threads per row
    const int row = tid >> 4, sub = tid & 15;
    float lmax = -1e30f;
    for (int c = sub; c < S_; c += 16) lmax = fmaxf(lmax, Ls[row][c]);
    red[row][sub] = lmax;
    __syncthreads();
    float mx = red[row][0];
#pragma unroll
    for (int t = 1; t < 16; ++t) mx = fmaxf(mx, red[row][t]);
    float lsum = 0.f;
    for (int c = sub; c < S_; c += 16) {
        float e = __expf(Ls[row][c] - mx);
        Ls[row][c] = e;
        lsum += e;
    }
    __syncthreads();
    red[row][sub] = lsum;
    __syncthreads();
    float s = 0.f;
#pragma unroll
    for (int t = 0; t < 16; ++t) s += red[row][t];
    float invsum = 1.0f / s;
    float* prow = probs + (size_t)(bh * S_ + iB + row) * S_;
    for (int c = sub; c < S_; c += 16) prow[c] = Ls[row][c] * invsum;
}

// ---- K4: ctx = probs @ v  (per head), f16 staging of probs ---------------
__global__ void __launch_bounds__(256)
ctx_gemm(const float* __restrict__ probs, const _Float16* __restrict__ vTh,
         _Float16* __restrict__ ctxh) {
    __shared__ _Float16 Ap[128][40];
    const int b = blockIdx.z, h = blockIdx.y, iBase = blockIdx.x * 128;
    const int tid = threadIdx.x, lane = tid & 31, wave = tid >> 5;
    const int lr = lane & 15, hh = lane >> 4;
    const int bh = b * H_ + h;
    v8f acc[6] = {};
    for (int kk = 0; kk < S_; kk += 32) {
        __syncthreads();
        for (int idx = tid; idx < 128 * 32; idx += 256) {
            int r = idx >> 5, c = idx & 31;
            Ap[r][c] = (_Float16)probs[(size_t)(bh * S_ + iBase + r) * S_ + kk + c];
        }
        __syncthreads();
        v16h a = frag_a(&Ap[wave * 16 + lr][0], hh);
#pragma unroll
        for (int t = 0; t < 6; ++t) {
            const _Float16* brow = vTh + (size_t)(bh * DH_ + t * 16 + lr) * S_ + kk;
            v16h bf = frag_b(brow, hh);
            acc[t] = wmma_f16(a, bf, acc[t]);
        }
    }
#pragma unroll
    for (int t = 0; t < 6; ++t) {
        int d = t * 16 + lr;
#pragma unroll
        for (int r = 0; r < 8; ++r) {
            int i = iBase + wave * 16 + r + 8 * hh;
            ctxh[(size_t)(b * S_ + i) * D_ + h * DH_ + d] = (_Float16)acc[t][r];
        }
    }
}

// ---- K5: output projection GEMM (fp32 out) -------------------------------
__global__ void __launch_bounds__(256)
gemm_proj(const _Float16* __restrict__ Ah, const _Float16* __restrict__ Wh,
          const float* __restrict__ bias, float* __restrict__ out) {
    __shared__ _Float16 As[32][72];
    __shared__ _Float16 Bs[64][72];
    const int mBase = blockIdx.y * 32;
    const int nBase = blockIdx.x * 64;
    const int tid = threadIdx.x, lane = tid & 31, wave = tid >> 5;
    const int wm = wave >> 2, wn = wave & 3;
    const int lr = lane & 15, hh = lane >> 4;
    v8f acc = {};
    for (int kk = 0; kk < D_; kk += 64) {
#pragma unroll
        for (int q3 = 0; q3 < 3; ++q3) {
            int c = tid + q3 * 256;
            const _Float16* gp;
            _Float16* lp;
            if (c < 256) {
                int r = c >> 3, col = (c & 7) * 8;
                gp = &Ah[(size_t)(mBase + r) * D_ + kk + col];
                lp = &As[r][col];
            } else {
                int c2 = c - 256;
                int r = c2 >> 3, col = (c2 & 7) * 8;
                gp = &Wh[(size_t)(nBase + r) * D_ + kk + col];
                lp = &Bs[r][col];
            }
            ASYNC_CP16(gp, lp);
        }
        ASYNC_WAIT();
        __syncthreads();
#pragma unroll
        for (int kz = 0; kz < 64; kz += 32) {
            v16h a = frag_a(&As[wm * 16 + lr][kz], hh);
            v16h b = frag_b(&Bs[wn * 16 + lr][kz], hh);
            acc = wmma_f16(a, b, acc);
        }
        __syncthreads();
    }
    const int n = nBase + wn * 16 + lr;
    const float bn = bias[n];
#pragma unroll
    for (int r = 0; r < 8; ++r) {
        int m = mBase + wm * 16 + r + 8 * hh;
        out[(size_t)m * D_ + n] = acc[r] + bn;
    }
}

// ---- K6: residual + LayerNorm --------------------------------------------
__global__ void __launch_bounds__(256)
ln_kernel(const float* __restrict__ proj, const float* __restrict__ hidden,
          const float* __restrict__ g, const float* __restrict__ beta,
          float* __restrict__ out) {
    __shared__ float rbuf[256];
    __shared__ float xs[D_];
    const int row = blockIdx.x, tid = threadIdx.x;
    float s = 0.f;
    for (int c = tid; c < D_; c += 256) {
        float x = proj[(size_t)row * D_ + c] + hidden[(size_t)row * D_ + c];
        xs[c] = x;
        s += x;
    }
    rbuf[tid] = s;
    __syncthreads();
    for (int st = 128; st > 0; st >>= 1) {
        if (tid < st) rbuf[tid] += rbuf[tid + st];
        __syncthreads();
    }
    float mu = rbuf[0] * (1.0f / D_);
    __syncthreads();
    float v = 0.f;
    for (int c = tid; c < D_; c += 256) { float d = xs[c] - mu; v += d * d; }
    rbuf[tid] = v;
    __syncthreads();
    for (int st = 128; st > 0; st >>= 1) {
        if (tid < st) rbuf[tid] += rbuf[tid + st];
        __syncthreads();
    }
    float var = rbuf[0] * (1.0f / D_);
    float inv = rsqrtf(var + LN_EPS);
    for (int c = tid; c < D_; c += 256)
        out[(size_t)row * D_ + c] = (xs[c] - mu) * inv * g[c] + beta[c];
}

// ---- launch ---------------------------------------------------------------
extern "C" void kernel_launch(void* const* d_in, const int* in_sizes, int n_in,
                              void* d_out, int out_size, void* d_ws, size_t ws_size,
                              hipStream_t stream) {
    (void)in_sizes; (void)n_in; (void)out_size; (void)ws_size;
    const float* hidden = (const float*)d_in[0];
    const float* dist   = (const float*)d_in[1];
    const float* ang    = (const float*)d_in[2];
    const float* mask   = (const float*)d_in[3];
    const float* Wq = (const float*)d_in[4];
    const float* bq = (const float*)d_in[5];
    const float* Wk = (const float*)d_in[6];
    const float* bk = (const float*)d_in[7];
    const float* Wv = (const float*)d_in[8];
    const float* bv = (const float*)d_in[9];
    const float* Wd = (const float*)d_in[10];
    const float* bd = (const float*)d_in[11];
    const float* Wa = (const float*)d_in[12];
    const float* ba = (const float*)d_in[13];
    const float* Wo = (const float*)d_in[14];
    const float* bo = (const float*)d_in[15];
    const float* ln_g = (const float*)d_in[16];
    const float* ln_b = (const float*)d_in[17];

    float* outp  = (float*)d_out;                          // [B,S,D]
    float* probs = outp + (size_t)B_ * S_ * D_;            // [B,H,S,S]

    char* ws = (char*)d_ws;
    size_t off = 0;
    auto alloc = [&](size_t bytes) -> void* {
        void* p = ws + off;
        off += (bytes + 255) & ~(size_t)255;
        return p;
    };
    const size_t NX = (size_t)B_ * S_ * D_;     // 1,179,648
    const size_t NW = (size_t)D_ * D_;          //   589,824
    const size_t NC = (size_t)B_ * H_ * S_;     //    12,288
    _Float16* Xh   = (_Float16*)alloc(NX * 2);
    _Float16* Wqh  = (_Float16*)alloc(NW * 2);
    _Float16* Wkh  = (_Float16*)alloc(NW * 2);
    _Float16* Wvh  = (_Float16*)alloc(NW * 2);
    _Float16* Woh  = (_Float16*)alloc(NW * 2);
    _Float16* qh   = (_Float16*)alloc(NX * 2);
    _Float16* kh   = (_Float16*)alloc(NX * 2);
    _Float16* vTh  = (_Float16*)alloc(NX * 2);
    _Float16* ctxh = (_Float16*)alloc(NX * 2);
    float*    cd   = (float*)alloc(NC * 4);
    float*    ca   = (float*)alloc(NC * 4);
    float*    cc   = (float*)alloc(NC * 4);
    float*    proj = (float*)alloc(NX * 4);

    // K0: conversions
    cvt_f32_f16<<<dim3((NX + 255) / 256), 256, 0, stream>>>(hidden, Xh, (int)NX);
    cvt_f32_f16<<<dim3((NW + 255) / 256), 256, 0, stream>>>(Wq, Wqh, (int)NW);
    cvt_f32_f16<<<dim3((NW + 255) / 256), 256, 0, stream>>>(Wk, Wkh, (int)NW);
    cvt_f32_f16<<<dim3((NW + 255) / 256), 256, 0, stream>>>(Wv, Wvh, (int)NW);
    cvt_f32_f16<<<dim3((NW + 255) / 256), 256, 0, stream>>>(Wo, Woh, (int)NW);

    // K1: QKV projections
    gemm_qkv<<<dim3(D_ / 64, (B_ * S_) / 32, 3), 256, 0, stream>>>(
        Xh, Wqh, Wkh, Wvh, bq, bk, bv, qh, kh, vTh);

    // K2: bias coefficients
    coeff_kernel<<<dim3((NC + 255) / 256), 256, 0, stream>>>(qh, Wd, bd, Wa, ba, cd, ca, cc);

    // K3: scores + softmax -> probs (fp32, directly in d_out)
    attn_scores<<<dim3(S_ / 16, H_, B_), 256, 0, stream>>>(
        qh, kh, dist, ang, mask, cd, ca, cc, probs);

    // K4: ctx = probs @ v
    ctx_gemm<<<dim3(S_ / 128, H_, B_), 256, 0, stream>>>(probs, vTh, ctxh);

    // K5: output projection
    gemm_proj<<<dim3(D_ / 64, (B_ * S_) / 32, 1), 256, 0, stream>>>(ctxh, Woh, bo, proj);

    // K6: residual + layernorm
    ln_kernel<<<dim3(B_ * S_), 256, 0, stream>>>(proj, hidden, ln_g, ln_b, outp);
}
